// GeLULayerForGATLayer_45105746542641
// MI455X (gfx1250) — matrix operand order repaired and checked by reading, not compiled, not run on hardware
//
#include <hip/hip_runtime.h>
#include <hip/hip_bf16.h>

// ---------------------------------------------------------------------------
// GAT (3 layers) for MI455X / gfx1250, wave32, WMMA f32_16x16x32_f16.
// N=20000 nodes (1250 16-row tiles), E=320000 edges, 128 features.
// edge_dst == repeat(arange(N),16) -> contiguous 16-edge segments per node.
// ---------------------------------------------------------------------------

typedef __attribute__((ext_vector_type(16))) _Float16 v16h;
typedef __attribute__((ext_vector_type(8)))  _Float16 v8h;
typedef __attribute__((ext_vector_type(8)))  float    v8f;

#define GAT_N 20000
#define GAT_E 320000

__device__ inline v8f wmma16(v16h a, v16h b, v8f c) {
  return __builtin_amdgcn_wmma_f32_16x16x32_f16(
      /*neg_a=*/false, a, /*neg_b=*/false, b,
      /*c_mod=*/(short)0, c, /*reuse_a=*/false, /*reuse_b=*/false);
}

// A-fragment (16x32 f16) per ISA layout: lane L holds row m=L%16, group g=L/16.
// Elements 0..7  <- k = koff..koff+7      (koff = kt*32 + g*8)
// Elements 8..15 <- k = koff+16..koff+23
__device__ inline v16h load_afrag(const _Float16* __restrict__ row, int koff) {
  v8h lo = *(const v8h*)(row + koff);
  v8h hi = *(const v8h*)(row + koff + 16);
  v16h a;
#pragma unroll
  for (int i = 0; i < 8; ++i) { a[i] = lo[i]; a[i + 8] = hi[i]; }
  return a;
}

// ---------------------------------------------------------------------------
// Repack weight W [P, Kreal] (row-major, f32) into fragment-ready f16 buffer:
// Bfrag[kt][ct][lane][i]  with  n = ct*16 + lane%16, k = kt*32 + (lane/16)*8 + (i%8) + (i/8)*16
// (zero-padded to Kpad). One aligned 32B load per lane per fragment at use.
// ---------------------------------------------------------------------------
__global__ void prep_bfrag(const float* __restrict__ W, _Float16* __restrict__ B,
                           int P, int Kreal, int Kpad) {
  int t = blockIdx.x * blockDim.x + threadIdx.x;
  int total = Kpad * P;
  if (t >= total) return;
  int i    = t & 15;
  int lane = (t >> 4) & 31;
  int Pt   = P >> 4;
  int ct   = (t >> 9) % Pt;
  int kt   = t / (512 * Pt);
  int n = ct * 16 + (lane & 15);
  int k = kt * 32 + (lane >> 4) * 8 + (i & 7) + (i >> 3) * 16;
  float v = (k < Kreal) ? W[(size_t)n * Kreal + k] : 0.0f;
  B[t] = (_Float16)v;
}

// ---------------------------------------------------------------------------
// Generic WMMA GEMM: C[M,P] = act(A[M,128] @ Bt + bias), K = 128 fixed.
// 4 waves/block, one 16-row tile per wave. ACT: 0=none, 1=leaky(0.2), 2=gelu.
// ---------------------------------------------------------------------------
template <int P, int ACT>
__global__ __launch_bounds__(128) void gemm_k128(
    const float* __restrict__ A, const _Float16* __restrict__ Bfrag,
    const float* __restrict__ bias, float* __restrict__ C, int M) {
  constexpr int PT = P / 16;
  __shared__ __align__(16) _Float16 As[64][136];  // 64 rows x 128, padded

  int tid = threadIdx.x;
  // Cooperative A load: 2 threads per row, f32 -> f16.
  {
    int r = tid >> 1, half = tid & 1;
    int gr = blockIdx.x * 64 + r;
    const float4* src = (gr < M) ? (const float4*)(A + (size_t)gr * 128 + half * 64)
                                 : nullptr;
#pragma unroll
    for (int j = 0; j < 16; ++j) {
      float4 v = make_float4(0.f, 0.f, 0.f, 0.f);
      if (src) v = src[j];
      _Float16* d = &As[r][half * 64 + j * 4];
      d[0] = (_Float16)v.x; d[1] = (_Float16)v.y;
      d[2] = (_Float16)v.z; d[3] = (_Float16)v.w;
    }
  }
  __syncthreads();

  int wave = tid >> 5, lane = tid & 31;
  int tile = blockIdx.x * 4 + wave;
  if (tile * 16 >= M) return;
  int m = lane & 15, g = lane >> 4;
  const _Float16* arow = &As[wave * 16 + m][0];
  int rbase = tile * 16 + g * 8;

  for (int ct = 0; ct < PT; ++ct) {
    v8f acc = {};
#pragma unroll
    for (int kt = 0; kt < 4; ++kt) {
      v16h a = load_afrag(arow, kt * 32 + g * 8);
      const _Float16* bp = Bfrag + (((size_t)(kt * PT + ct) * 32 + lane) << 4);
      v16h b = *(const v16h*)bp;
      acc = wmma16(a, b, acc);
    }
    int n = ct * 16 + m;
    float bv = bias[n];
#pragma unroll
    for (int v = 0; v < 8; ++v) {
      float x = acc[v] + bv;
      if (ACT == 1) x = x > 0.f ? x : 0.2f * x;
      else if (ACT == 2) x = 0.5f * x * (1.0f + erff(x * 0.70710678118654752f));
      C[(size_t)(rbase + v) * P + n] = x;
    }
  }
}

// ---------------------------------------------------------------------------
// Fused V1 edge kernel: nbr = lrelu(fc2 @ [atom[src]||bond]), then
// logits[e] = lrelu(att1.new_atom[dst] (=alpha) + att2.nbr + att_b).
// Never materializes nbr. 4 waves/block, one 16-edge tile per wave.
// K = 144 padded to 160 (5 k-tiles of 32).
// ---------------------------------------------------------------------------
__global__ __launch_bounds__(128) void v1_edge_logits(
    const float* __restrict__ atom, const int* __restrict__ esrc,
    const int* __restrict__ edst, const float* __restrict__ bond,
    const _Float16* __restrict__ Bfrag, const float* __restrict__ b2,
    const float* __restrict__ attw, const float* __restrict__ attb,
    const float* __restrict__ alpha, float* __restrict__ logits) {
  __shared__ __align__(16) _Float16 As[4][16][168];
  __shared__ float red[4][32][8];

  int tid = threadIdx.x, wave = tid >> 5, lane = tid & 31;
  int tile = blockIdx.x * 4 + wave;  // 20000 tiles total, grid exact
  int e0 = tile * 16;
  int r = lane & 15, g = lane >> 4;
  int e = e0 + r;
  int src = esrc[e];

  {  // atom part: this lane fills k = g*64 .. g*64+63 of row r
    const float4* ap = (const float4*)(atom + (size_t)src * 128 + g * 64);
#pragma unroll
    for (int j = 0; j < 16; ++j) {
      float4 v = ap[j];
      _Float16* d = &As[wave][r][g * 64 + j * 4];
      d[0] = (_Float16)v.x; d[1] = (_Float16)v.y;
      d[2] = (_Float16)v.z; d[3] = (_Float16)v.w;
    }
  }
  if (g == 0) {  // bond part: k = 128..143
    const float4* bp = (const float4*)(bond + (size_t)e * 16);
#pragma unroll
    for (int j = 0; j < 4; ++j) {
      float4 v = bp[j];
      _Float16* d = &As[wave][r][128 + j * 4];
      d[0] = (_Float16)v.x; d[1] = (_Float16)v.y;
      d[2] = (_Float16)v.z; d[3] = (_Float16)v.w;
    }
  } else {  // zero pad k = 144..159
#pragma unroll
    for (int k = 144; k < 160; ++k) As[wave][r][k] = (_Float16)0.f;
  }
  __syncthreads();

  int m = lane & 15;
  const _Float16* arow = &As[wave][m][0];
  float part[8];
#pragma unroll
  for (int v = 0; v < 8; ++v) part[v] = 0.f;

  for (int ct = 0; ct < 8; ++ct) {
    v8f acc = {};
#pragma unroll
    for (int kt = 0; kt < 5; ++kt) {
      v16h a = load_afrag(arow, kt * 32 + g * 8);
      const _Float16* bp = Bfrag + (((size_t)(kt * 8 + ct) * 32 + lane) << 4);
      v16h b = *(const v16h*)bp;
      acc = wmma16(a, b, acc);
    }
    int n = ct * 16 + m;
    float bv = b2[n];
    float aw = attw[128 + n];
#pragma unroll
    for (int v = 0; v < 8; ++v) {
      float x = acc[v] + bv;
      x = x > 0.f ? x : 0.2f * x;       // lrelu on nbr
      part[v] += x * aw;                // dot with att2
    }
  }
#pragma unroll
  for (int v = 0; v < 8; ++v) red[wave][lane][v] = part[v];
  __syncthreads();

  if (lane < 16) {  // edge row m: sum 16 lane-partials
    int mm = lane;
    float s = 0.f;
#pragma unroll
    for (int j = 0; j < 16; ++j) s += red[wave][(mm >> 3) * 16 + j][mm & 7];
    int ee = e0 + mm;
    float lg = alpha[edst[ee]] + s + attb[0];
    logits[ee] = lg > 0.f ? lg : 0.2f * lg;
  }
}

// Per-node dot products: o1[n] = w1 . X[n], o2[n] = w2 . X[n] (w2 optional).
__global__ __launch_bounds__(128) void node_dot2(
    const float* __restrict__ X, const float* __restrict__ w1,
    const float* __restrict__ w2, float* __restrict__ o1,
    float* __restrict__ o2, int Nn) {
  int wave = threadIdx.x >> 5, lane = threadIdx.x & 31;
  int n = blockIdx.x * 4 + wave;
  if (n >= Nn) return;
  const float* x = X + (size_t)n * 128;
  float s1 = 0.f, s2 = 0.f;
#pragma unroll
  for (int j = 0; j < 4; ++j) {
    float xv = x[lane + 32 * j];
    s1 += w1[lane + 32 * j] * xv;
    if (w2) s2 += w2[lane + 32 * j] * xv;
  }
#pragma unroll
  for (int off = 16; off; off >>= 1) {
    s1 += __shfl_xor(s1, off, 32);
    if (w2) s2 += __shfl_xor(s2, off, 32);
  }
  if (lane == 0) { o1[n] = s1; if (w2) o2[n] = s2; }
}

// V2 edge logits: lg[e] = lrelu(d1[dst] + d2[src] + b)
__global__ void v2_edge_logits(const int* __restrict__ esrc,
                               const int* __restrict__ edst,
                               const float* __restrict__ d1,
                               const float* __restrict__ d2,
                               const float* __restrict__ b,
                               float* __restrict__ logits, int E) {
  int e = blockIdx.x * blockDim.x + threadIdx.x;
  if (e >= E) return;
  float lg = d1[edst[e]] + d2[esrc[e]] + b[0];
  logits[e] = lg > 0.f ? lg : 0.2f * lg;
}

// Segment softmax (contiguous 16-edge segments) + weighted gather + ELU.
__global__ __launch_bounds__(128) void attn_aggregate(
    const float* __restrict__ logits, const int* __restrict__ esrc,
    const float* __restrict__ nt, float* __restrict__ ctx) {
  __shared__ float lg[16];
  __shared__ int ss[16];
  int i = blockIdx.x, f = threadIdx.x;
  if (f < 16) { lg[f] = logits[i * 16 + f]; ss[f] = esrc[i * 16 + f]; }
  __syncthreads();
  float mx = lg[0];
#pragma unroll
  for (int e = 1; e < 16; ++e) mx = fmaxf(mx, lg[e]);
  float den = 0.f, acc = 0.f;
#pragma unroll
  for (int e = 0; e < 16; ++e) {
    float w = __expf(lg[e] - mx);
    den += w;
    acc += w * nt[(size_t)ss[e] * 128 + f];
  }
  float t = acc / den;
  ctx[(size_t)i * 128 + f] = t > 0.f ? t : (__expf(t) - 1.0f);  // ELU(alpha=1)
}

// GRU gate combine: hn = (1-z)*tanh(i_n + r*h_n) + z*h
__global__ void gru_combine(const float* __restrict__ gi,
                            const float* __restrict__ gh,
                            const float* __restrict__ h,
                            float* __restrict__ hn, int Nn) {
  int idx = blockIdx.x * blockDim.x + threadIdx.x;
  if (idx >= Nn * 128) return;
  int n = idx >> 7, f = idx & 127;
  size_t base = (size_t)n * 384 + f;
  float ir = gi[base], iz = gi[base + 128], inn = gi[base + 256];
  float hr = gh[base], hz = gh[base + 128], hnn = gh[base + 256];
  float r = 1.f / (1.f + __expf(-(ir + hr)));
  float z = 1.f / (1.f + __expf(-(iz + hz)));
  float nv = tanhf(inn + r * hnn);
  hn[idx] = (1.f - z) * nv + z * h[idx];
}

__global__ void avg3(const float* __restrict__ a, const float* __restrict__ b,
                     const float* __restrict__ c, float* __restrict__ o, int Nn) {
  int i = blockIdx.x * blockDim.x + threadIdx.x;
  if (i < Nn * 128) o[i] = (a[i] + b[i] + c[i]) * (1.0f / 3.0f);
}

// ---------------------------------------------------------------------------
extern "C" void kernel_launch(void* const* d_in, const int* in_sizes, int n_in,
                              void* d_out, int out_size, void* d_ws, size_t ws_size,
                              hipStream_t stream) {
  const int N = GAT_N, E = GAT_E;
  const float* atom   = (const float*)d_in[0];
  const int*   esrc   = (const int*)d_in[1];
  const int*   edst   = (const int*)d_in[2];
  const float* bond   = (const float*)d_in[3];
  const float* fc1_w  = (const float*)d_in[4];
  const float* fc1_b  = (const float*)d_in[5];
  const float* fc2_w  = (const float*)d_in[6];
  const float* fc2_b  = (const float*)d_in[7];
  const float* att_w  = (const float*)d_in[8];
  const float* att_b  = (const float*)d_in[9];
  const float* atnd_w = (const float*)d_in[10];
  const float* atnd_b = (const float*)d_in[11];
  const float* wih1   = (const float*)d_in[12];
  const float* whh1   = (const float*)d_in[13];
  const float* bih1   = (const float*)d_in[14];
  const float* bhh1   = (const float*)d_in[15];
  const float* fc1v2_w = (const float*)d_in[16];
  const float* fc1v2_b = (const float*)d_in[17];
  const float* fc2v2_w = (const float*)d_in[18];
  const float* fc2v2_b = (const float*)d_in[19];
  const float* wih2   = (const float*)d_in[20];
  const float* whh2   = (const float*)d_in[21];
  const float* bih2   = (const float*)d_in[22];
  const float* bhh2   = (const float*)d_in[23];
  const float* lin_w  = (const float*)d_in[24];
  const float* lin_b  = (const float*)d_in[25];

  // d_out layout: output | all_stack(h1,h2,h3) | G_0 | averaged  (f32 each N*128)
  float* out = (float*)d_out;
  size_t SZ = (size_t)N * 128;
  float* h1  = out + SZ;
  float* h2  = out + 2 * SZ;
  float* h3  = out + 3 * SZ;
  float* G0  = out + 4 * SZ;
  float* avg = out + 5 * SZ;

  // Workspace carve-up (256B aligned).
  char* ws = (char*)d_ws;
  size_t off = 0;
  auto alloc = [&](size_t bytes) -> void* {
    void* p = ws + off;
    off = (off + bytes + 255) & ~(size_t)255;
    return p;
  };
  _Float16* bf_fc1   = (_Float16*)alloc((size_t)128 * 128 * 2);
  _Float16* bf_atnd  = (_Float16*)alloc((size_t)128 * 128 * 2);
  _Float16* bf_fc2   = (_Float16*)alloc((size_t)160 * 128 * 2);
  _Float16* bf_wih1  = (_Float16*)alloc((size_t)128 * 384 * 2);
  _Float16* bf_whh1  = (_Float16*)alloc((size_t)128 * 384 * 2);
  _Float16* bf_fc2v2 = (_Float16*)alloc((size_t)128 * 128 * 2);
  _Float16* bf_wih2  = (_Float16*)alloc((size_t)128 * 384 * 2);
  _Float16* bf_whh2  = (_Float16*)alloc((size_t)128 * 384 * 2);
  _Float16* bf_lin   = (_Float16*)alloc((size_t)128 * 128 * 2);
  float* nt   = (float*)alloc(SZ * 4);
  float* ctxb = (float*)alloc(SZ * 4);
  float* gi   = (float*)alloc((size_t)N * 384 * 4);
  float* gh   = (float*)alloc((size_t)N * 384 * 4);
  float* lgts = (float*)alloc((size_t)E * 4);
  float* d1   = (float*)alloc((size_t)N * 4);
  float* d2   = (float*)alloc((size_t)N * 4);

  // ---- Weight repack (f32 -> fragment-ready f16) ----
  prep_bfrag<<<64, 256, 0, stream>>>(fc1_w, bf_fc1, 128, 128, 128);
  prep_bfrag<<<64, 256, 0, stream>>>(atnd_w, bf_atnd, 128, 128, 128);
  prep_bfrag<<<80, 256, 0, stream>>>(fc2_w, bf_fc2, 128, 144, 160);
  prep_bfrag<<<192, 256, 0, stream>>>(wih1, bf_wih1, 384, 128, 128);
  prep_bfrag<<<192, 256, 0, stream>>>(whh1, bf_whh1, 384, 128, 128);
  prep_bfrag<<<64, 256, 0, stream>>>(fc2v2_w, bf_fc2v2, 128, 128, 128);
  prep_bfrag<<<192, 256, 0, stream>>>(wih2, bf_wih2, 384, 128, 128);
  prep_bfrag<<<192, 256, 0, stream>>>(whh2, bf_whh2, 384, 128, 128);
  prep_bfrag<<<64, 256, 0, stream>>>(lin_w, bf_lin, 128, 128, 128);

  const int gemmGrid = (1250 + 3) / 4;  // 313 blocks of 4 row-tile waves

  // ---- GATlayerV1 ----
  gemm_k128<128, 1><<<gemmGrid, 128, 0, stream>>>(atom, bf_fc1, fc1_b, G0, N);
  gemm_k128<128, 0><<<gemmGrid, 128, 0, stream>>>(G0, bf_atnd, atnd_b, nt, N);
  node_dot2<<<5000, 128, 0, stream>>>(G0, att_w, nullptr, d1, nullptr, N);
  v1_edge_logits<<<5000, 128, 0, stream>>>(atom, esrc, edst, bond, bf_fc2,
                                           fc2_b, att_w, att_b, d1, lgts);
  attn_aggregate<<<N, 128, 0, stream>>>(lgts, esrc, nt, ctxb);
  gemm_k128<384, 0><<<gemmGrid, 128, 0, stream>>>(ctxb, bf_wih1, bih1, gi, N);
  gemm_k128<384, 0><<<gemmGrid, 128, 0, stream>>>(G0, bf_whh1, bhh1, gh, N);
  gru_combine<<<10000, 256, 0, stream>>>(gi, gh, G0, h1, N);

  // ---- GATlayerV2 x2 (shared weights) ----
  const float* hin = h1;
  float* houts[2] = {h2, h3};
  for (int l = 0; l < 2; ++l) {
    gemm_k128<128, 0><<<gemmGrid, 128, 0, stream>>>(hin, bf_fc2v2, fc2v2_b, nt, N);
    node_dot2<<<5000, 128, 0, stream>>>(hin, fc1v2_w, fc1v2_w + 128, d1, d2, N);
    v2_edge_logits<<<1250, 256, 0, stream>>>(esrc, edst, d1, d2, fc1v2_b, lgts, E);
    attn_aggregate<<<N, 128, 0, stream>>>(lgts, esrc, nt, ctxb);
    gemm_k128<384, 0><<<gemmGrid, 128, 0, stream>>>(ctxb, bf_wih2, bih2, gi, N);
    gemm_k128<384, 0><<<gemmGrid, 128, 0, stream>>>(hin, bf_whh2, bhh2, gh, N);
    gru_combine<<<10000, 256, 0, stream>>>(gi, gh, hin, houts[l], N);
    hin = houts[l];
  }

  // ---- Readout ----
  avg3<<<10000, 256, 0, stream>>>(h1, h2, h3, avg, N);
  gemm_k128<128, 2><<<gemmGrid, 128, 0, stream>>>(avg, bf_lin, lin_b, out, N);
}
